// GraphCodeBERT_RR_17952963297678
// MI455X (gfx1250) — compile-verified, matrix operand order checked by reading, not run on hardware
//
#include <hip/hip_runtime.h>
#include <hip/hip_bf16.h>
#include <math.h>

// ---------------------------------------------------------------------------
// GraphCodeBERT head, collapsed: only emb[:,0] feeds the output, so we compute
// per-batch masked mean of gathered embedding rows, then two 64x768x768 WMMA
// GEMMs (tanh / gelu epilogues), then LayerNorm + dot + tanh.
// ---------------------------------------------------------------------------

#define B_ 64
#define N_ 640
#define H_ 768

typedef __attribute__((ext_vector_type(16))) _Float16 v16h;
typedef __attribute__((ext_vector_type(8)))  _Float16 v8h;
typedef __attribute__((ext_vector_type(8)))  float    v8f;

// ---------------------------------------------------------------------------
// Kernel 1: per-batch gather + masked average of embedding rows.
// One block per batch (64 blocks x 256 threads). Thread t owns columns
// {t, t+256, t+512} -> fully coalesced 768-float row reads. The flag loop is
// block-uniform (flags in LDS), so no divergence while streaming ~126 MB.
// Output written as f16 (A-matrix of GEMM1).
// ---------------------------------------------------------------------------
__global__ void gcb_gather_avg(const int* __restrict__ inputs,
                               const int* __restrict__ pos,
                               const unsigned char* __restrict__ amask, // [B,N,N] bool
                               const float* __restrict__ emb,           // [V,H]
                               _Float16* __restrict__ e0h) {            // [B,H] f16
    const int b = blockIdx.x;
    const int t = threadIdx.x;

    __shared__ unsigned char s_flag[N_];
    __shared__ int           s_row[N_];
    __shared__ int           s_cnt;
    if (t == 0) s_cnt = 0;
    __syncthreads();

    int local = 0;
    for (int mm = t; mm < N_; mm += 256) {
        const bool f = (pos[b * N_ + mm] >= 2) &&
                       (amask[(size_t)b * N_ * N_ + mm] != 0); // attn row 0
        s_flag[mm] = (unsigned char)f;
        s_row[mm]  = inputs[b * N_ + mm];
        local += f ? 1 : 0;
    }
    if (local) atomicAdd(&s_cnt, local);
    __syncthreads();

    const bool node0 = (pos[b * N_] == 0);
    float a0 = 0.f, a1 = 0.f, a2 = 0.f;

    if (node0) {
        for (int mm = 0; mm < N_; ++mm) {
            if (s_flag[mm]) {
                const float* __restrict__ r = emb + (size_t)s_row[mm] * H_;
                a0 += r[t];
                a1 += r[t + 256];
                a2 += r[t + 512];
            }
        }
        const float sc = 1.0f / ((float)s_cnt + 1e-10f);
        a0 *= sc; a1 *= sc; a2 *= sc;
    } else {
        const float* __restrict__ r = emb + (size_t)s_row[0] * H_;
        a0 = r[t]; a1 = r[t + 256]; a2 = r[t + 512];
    }

    _Float16* o = e0h + (size_t)b * H_;
    o[t]       = (_Float16)a0;
    o[t + 256] = (_Float16)a1;
    o[t + 512] = (_Float16)a2;
}

// ---------------------------------------------------------------------------
// Kernel 2/3: WMMA GEMM  C[64,768] = act(A[64,768] @ W[768,768] + bias)
// One 16x16 output tile per wave; K consumed 32 at a time with
// v_wmma_f32_16x16x32_f16. A is f16 row-major (16B vector loads). W is f32
// row-major; the B-operand loader does the strided column reads and converts
// f32->f16 inline (GEMM is ~38 MFLOP, pure latency-trivial).
// mode 0: tanh epilogue, f16 store (feeds next GEMM's A)
// mode 1: exact gelu epilogue, f32 store (feeds LayerNorm)
// Grid: 24 blocks x 256 threads = 192 waves = (64/16)*(768/16) tiles.
// ---------------------------------------------------------------------------
__global__ void gcb_wmma_gemm(const _Float16* __restrict__ A,
                              const float* __restrict__ W,
                              const float* __restrict__ bias,
                              void* __restrict__ outp,
                              int mode) {
    const int K  = H_;
    const int NN = H_;
    const int wave = (blockIdx.x * blockDim.x + threadIdx.x) >> 5; // 0..191
    const int lane = threadIdx.x & 31;
    const int tn = wave % (H_ / 16);   // 0..47
    const int tm = wave / (H_ / 16);   // 0..3
    const int m0 = tm * 16;
    const int n0 = tn * 16;

    const int mr    = m0 + (lane & 15);        // A row for this lane
    const int n     = n0 + (lane & 15);        // B/C column for this lane
    const int khalf = (lane >> 4) * 8;         // K-half selector (ISA layout)

    v8f c = {};
    for (int k0 = 0; k0 < K; k0 += 32) {
        // A operand: lane holds K in {khalf..khalf+7} U {khalf+16..khalf+23}
        const v8h alo = *(const v8h*)(A + (size_t)mr * K + k0 + khalf);
        const v8h ahi = *(const v8h*)(A + (size_t)mr * K + k0 + 16 + khalf);
        v16h a, bm;
#pragma unroll
        for (int i = 0; i < 8; ++i) { a[i] = alo[i]; a[8 + i] = ahi[i]; }

        // B operand: column n, same K split; f32 -> f16 inline
        const float* __restrict__ wp0 = W + (size_t)(k0 + khalf) * NN + n;
        const float* __restrict__ wp1 = W + (size_t)(k0 + 16 + khalf) * NN + n;
#pragma unroll
        for (int i = 0; i < 8; ++i) {
            bm[i]     = (_Float16)wp0[(size_t)i * NN];
            bm[8 + i] = (_Float16)wp1[(size_t)i * NN];
        }

        c = __builtin_amdgcn_wmma_f32_16x16x32_f16(
                /*neg_a=*/false, a, /*neg_b=*/false, bm,
                /*c_mod=*/(short)0, c, /*reuse_a=*/false, /*reuse_b=*/false);
    }

    // C layout: VGPR j -> row m0 + 8*(lane>>4) + j, col n
    const float bv    = bias[n];
    const int   mbase = m0 + (lane >> 4) * 8;
    if (mode == 0) {
        _Float16* out = (_Float16*)outp;
#pragma unroll
        for (int j = 0; j < 8; ++j)
            out[(size_t)(mbase + j) * NN + n] = (_Float16)tanhf(c[j] + bv);
    } else {
        float* out = (float*)outp;
#pragma unroll
        for (int j = 0; j < 8; ++j) {
            const float x = c[j] + bv;
            out[(size_t)(mbase + j) * NN + n] =
                0.5f * x * (1.0f + erff(x * 0.70710678118654752f));
        }
    }
}

// ---------------------------------------------------------------------------
// Kernel 4: per-row LayerNorm + dot(dec_w) + tanh. One block per batch row.
// ---------------------------------------------------------------------------
__global__ void gcb_ln_dot(const float* __restrict__ X,   // [B,H]
                           const float* __restrict__ g,
                           const float* __restrict__ bta,
                           const float* __restrict__ dw,  // [H]
                           const float* __restrict__ db,  // [1]
                           float* __restrict__ out) {     // [B]
    const int b = blockIdx.x;
    const int t = threadIdx.x;
    const float* __restrict__ x = X + (size_t)b * H_;

    const float v0 = x[t], v1 = x[t + 256], v2 = x[t + 512];

    __shared__ float red[256];
    red[t] = v0 + v1 + v2;
    __syncthreads();
    for (int off = 128; off > 0; off >>= 1) {
        if (t < off) red[t] += red[t + off];
        __syncthreads();
    }
    const float mu = red[0] * (1.0f / H_);
    __syncthreads();

    const float d0 = v0 - mu, d1 = v1 - mu, d2 = v2 - mu;
    red[t] = d0 * d0 + d1 * d1 + d2 * d2;
    __syncthreads();
    for (int off = 128; off > 0; off >>= 1) {
        if (t < off) red[t] += red[t + off];
        __syncthreads();
    }
    const float rs = rsqrtf(red[0] * (1.0f / H_) + 1e-5f);
    __syncthreads();

    red[t] = (d0 * rs * g[t]       + bta[t])       * dw[t]
           + (d1 * rs * g[t + 256] + bta[t + 256]) * dw[t + 256]
           + (d2 * rs * g[t + 512] + bta[t + 512]) * dw[t + 512];
    __syncthreads();
    for (int off = 128; off > 0; off >>= 1) {
        if (t < off) red[t] += red[t + off];
        __syncthreads();
    }
    if (t == 0) out[b] = tanhf(red[0] + db[0]);
}

// ---------------------------------------------------------------------------
extern "C" void kernel_launch(void* const* d_in, const int* in_sizes, int n_in,
                              void* d_out, int out_size, void* d_ws, size_t ws_size,
                              hipStream_t stream) {
    (void)in_sizes; (void)n_in; (void)out_size; (void)ws_size;

    const int*           inputs   = (const int*)d_in[0];
    const int*           pos      = (const int*)d_in[1];
    const unsigned char* amask    = (const unsigned char*)d_in[2]; // bool
    const float*         emb      = (const float*)d_in[3];
    const float*         pooler_w = (const float*)d_in[4];
    const float*         pooler_b = (const float*)d_in[5];
    const float*         dense_w  = (const float*)d_in[6];
    const float*         dense_b  = (const float*)d_in[7];
    const float*         ln_g     = (const float*)d_in[8];
    const float*         ln_b     = (const float*)d_in[9];
    const float*         dec_w    = (const float*)d_in[10];
    const float*         dec_b    = (const float*)d_in[11];
    float*               out      = (float*)d_out;

    // workspace layout (all offsets 256B-aligned)
    char* ws = (char*)d_ws;
    _Float16* e0h     = (_Float16*)(ws);                         //  96 KB
    _Float16* pooledh = (_Float16*)(ws + (size_t)B_ * H_ * 2);   //  96 KB
    float*    xf      = (float*)  (ws + (size_t)B_ * H_ * 4);    // 192 KB

    // 1) gather + masked average -> e0 (f16)
    gcb_gather_avg<<<B_, 256, 0, stream>>>(inputs, pos, amask, emb, e0h);

    // 2) pooled = tanh(e0 @ pooler_w + b)   (f16 out)
    gcb_wmma_gemm<<<24, 256, 0, stream>>>(e0h, pooler_w, pooler_b,
                                          (void*)pooledh, /*mode=*/0);

    // 3) x = gelu(pooled @ dense_w + b)     (f32 out)
    gcb_wmma_gemm<<<24, 256, 0, stream>>>(pooledh, dense_w, dense_b,
                                          (void*)xf, /*mode=*/1);

    // 4) out = tanh(LN(x) . dec_w + dec_b)
    gcb_ln_dot<<<B_, 256, 0, stream>>>(xf, ln_g, ln_b, dec_w, dec_b, out);
}
